// Attention_49417893707810
// MI455X (gfx1250) — matrix operand order repaired and checked by reading, not compiled, not run on hardware
//
#include <hip/hip_runtime.h>

// MI455X / gfx1250 fused attention:
//   cvt(f32->bf16) -> QKV gemm + fused RoPE (bf16 WMMA) -> flash attention
//   (bf16 WMMA, f32 online softmax) -> output projection (bf16 WMMA, f32 out)
// Compute-bound workload (~206 GFLOP vs ~130MB data): everything routed
// through v_wmma_f32_16x16x32_bf16, fp32 accumulation throughout.

typedef __bf16 bf16;
typedef __attribute__((ext_vector_type(16))) __bf16 v16bf;
typedef __attribute__((ext_vector_type(8)))  __bf16 v8bf;
typedef __attribute__((ext_vector_type(4)))  __bf16 v4bf;
typedef __attribute__((ext_vector_type(8)))  float  v8f;

#define B_  2
#define S_  2048
#define D_  2048
#define H_  16
#define HD_ 128
#define M_  (B_ * S_)   // 4096 flattened tokens

static __device__ __forceinline__ v16bf merge16(v8bf lo, v8bf hi) {
  union { v16bf v; v8bf h[2]; } u;
  u.h[0] = lo; u.h[1] = hi;
  return u.v;
}

static __device__ __forceinline__ v8f wmma_bf16(v16bf a, v16bf b, v8f c) {
  // (neg_a, A, neg_b, B, c_mod, C, reuse_a, reuse_b)
  return __builtin_amdgcn_wmma_f32_16x16x32_bf16(false, a, false, b, (short)0, c,
                                                 false, false);
}

// A operand: 16(M) x 32(K) bf16 tile from row-major source (row stride ld).
// lanes 0-15: M=lane, K = {0..7, 16..23}; lanes 16-31: M=lane-16, K = {8..15, 24..31}
static __device__ __forceinline__ v16bf load_a_frag(const bf16* base, int ld, int lane) {
  const int row = lane & 15;
  const int ko  = (lane >> 4) * 8;
  v8bf lo = *(const v8bf*)(base + row * ld + ko);
  v8bf hi = *(const v8bf*)(base + row * ld + ko + 16);
  return merge16(lo, hi);
}

// B operand: 32(K) x 16(N); source row-major indexed by n, contiguous in k
// (i.e. B^T rows). lane n = lane&15; lanes 0-15: K = {0..7, 16..23},
// lanes 16-31: K = {8..15, 24..31}.
static __device__ __forceinline__ v16bf load_b_frag(const bf16* base, int ld, int lane) {
  const int n  = lane & 15;
  const int ks = (lane >> 4) * 8;
  v8bf lo = *(const v8bf*)(base + n * ld + ks);
  v8bf hi = *(const v8bf*)(base + n * ld + ks + 16);
  return merge16(lo, hi);
}

// ---------------------------------------------------------------- converters
__global__ void cvt_f32_bf16(const float* __restrict__ in, bf16* __restrict__ out,
                             int n4) {
  int i = blockIdx.x * blockDim.x + threadIdx.x;
  if (i < n4) {
    float4 f = ((const float4*)in)[i];
    v4bf o;
    o[0] = (bf16)f.x; o[1] = (bf16)f.y; o[2] = (bf16)f.z; o[3] = (bf16)f.w;
    ((v4bf*)out)[i] = o;
  }
}

// ------------------------------------------------- QKV projection + RoPE
// C = X(M x D) @ W^T (W is N x K row-major -> B-fragment friendly).
// z = 0: Q (RoPE, store [bh][s][hd]); z = 1: K (RoPE, same layout);
// z = 2: V (store transposed [bh][hd][s] for the P*V B-operand).
__global__ __launch_bounds__(256)
void qkv_rope_gemm(const bf16* __restrict__ X,
                   const bf16* __restrict__ WQ,
                   const bf16* __restrict__ WK,
                   const bf16* __restrict__ WV,
                   bf16* __restrict__ Q, bf16* __restrict__ K,
                   bf16* __restrict__ Vt) {
  __shared__ bf16 As[128 * 32];
  __shared__ bf16 Bs[128 * 32];
  const int z = blockIdx.z;
  const bf16* W = (z == 0) ? WQ : (z == 1) ? WK : WV;
  const int tileM = blockIdx.x * 128;
  const int tileN = blockIdx.y * 128;
  const int tid  = threadIdx.x;
  const int wave = tid >> 5, lane = tid & 31;
  const int wm = (wave & 3) * 32;   // 4 waves along M (2x 16-row tiles each)
  const int wn = (wave >> 2) * 64;  // 2 waves along N (4x 16-col tiles each)
  const int half = lane >> 4;

  v8f acc[2][4];
#pragma unroll
  for (int a = 0; a < 2; ++a)
#pragma unroll
    for (int b = 0; b < 4; ++b)
#pragma unroll
      for (int r = 0; r < 8; ++r) acc[a][b][r] = 0.f;

  for (int k0 = 0; k0 < D_; k0 += 32) {
#pragma unroll
    for (int part = 0; part < 2; ++part) {   // 512 x 16B chunks per tile
      int idx = tid + part * 256;
      int row = idx >> 2, c = (idx & 3) * 8;
      *(v8bf*)(As + row * 32 + c) = *(const v8bf*)(X + (size_t)(tileM + row) * D_ + k0 + c);
      *(v8bf*)(Bs + row * 32 + c) = *(const v8bf*)(W + (size_t)(tileN + row) * D_ + k0 + c);
    }
    __syncthreads();
    v16bf af0 = load_a_frag(As + (wm +  0) * 32, 32, lane);
    v16bf af1 = load_a_frag(As + (wm + 16) * 32, 32, lane);
#pragma unroll
    for (int nt = 0; nt < 4; ++nt) {
      v16bf bf = load_b_frag(Bs + (wn + nt * 16) * 32, 32, lane);
      acc[0][nt] = wmma_bf16(af0, bf, acc[0][nt]);
      acc[1][nt] = wmma_bf16(af1, bf, acc[1][nt]);
    }
    __syncthreads();
  }

  // Epilogue: C/D layout -> lane holds column (lane&15), VGPR r holds row
  // r + 8*(lane>>4). RoPE pairs adjacent channels = adjacent lanes.
#pragma unroll
  for (int mt = 0; mt < 2; ++mt) {
#pragma unroll
    for (int nt = 0; nt < 4; ++nt) {
      const int col = tileN + wn + nt * 16 + (lane & 15);
      const int h = col >> 7, dh = col & 127, d2 = dh >> 1;
#pragma unroll
      for (int r = 0; r < 8; ++r) {
        const int row = tileM + wm + mt * 16 + r + 8 * half;
        const int b = row >> 11, pos = row & (S_ - 1);
        float v = acc[mt][nt][r];
        if (z < 2) {
          // inv_freq = 10000^(-2*d2/128) = exp2(-log2(1e4)/64 * d2)
          float th = (float)pos * exp2f(-0.20762050593046f * (float)d2);
          float sn, cs;
          __sincosf(th, &sn, &cs);
          float p = __shfl_xor(v, 1, 32);  // partner channel (xr <-> xi)
          v = (lane & 1) ? (p * sn + v * cs)   // odd:  xr*s + xi*c
                         : (v * cs - p * sn);  // even: xr*c - xi*s
          bf16* dst = (z == 0) ? Q : K;
          dst[((size_t)(b * H_ + h) * S_ + pos) * HD_ + dh] = (bf16)v;
        } else {
          Vt[((size_t)(b * H_ + h) * HD_ + dh) * S_ + pos] = (bf16)v;
        }
      }
    }
  }
}

// ------------------------------------------------------- flash attention
// One wave = one 16-row query tile; online softmax over 32-key blocks.
__global__ __launch_bounds__(256)
void flash_attn(const bf16* __restrict__ Q, const bf16* __restrict__ K,
                const bf16* __restrict__ Vt, bf16* __restrict__ ctx) {
  __shared__ bf16 P[8 * 16 * 32];  // per-wave 16x32 probability staging
  const int tid = threadIdx.x, wave = tid >> 5, lane = tid & 31;
  const int bh = blockIdx.y, b = bh >> 4, h = bh & 15;
  const int q0 = blockIdx.x * 128 + wave * 16;
  const int half = lane >> 4;
  const bf16* Qb = Q  + (size_t)bh * S_ * HD_;
  const bf16* Kb = K  + (size_t)bh * S_ * HD_;
  const bf16* Vb = Vt + (size_t)bh * HD_ * S_;
  bf16* pl = P + wave * (16 * 32);

  v16bf qf[4];
#pragma unroll
  for (int kk = 0; kk < 4; ++kk)
    qf[kk] = load_a_frag(Qb + (size_t)q0 * HD_ + kk * 32, HD_, lane);

  v8f O[8];
#pragma unroll
  for (int c = 0; c < 8; ++c)
#pragma unroll
    for (int r = 0; r < 8; ++r) O[c][r] = 0.f;
  float mrow[8], lrow[8];
#pragma unroll
  for (int r = 0; r < 8; ++r) { mrow[r] = -1e30f; lrow[r] = 0.f; }

  const float scale = 0.08838834764831845f;  // 1/sqrt(128)
  const int nkb = (q0 + 47) >> 5;            // 32-key blocks up to diag
  for (int kb = 0; kb < nkb; ++kb) {
    v8f sc[2];
#pragma unroll
    for (int t = 0; t < 2; ++t) {
      v8f s;
#pragma unroll
      for (int r = 0; r < 8; ++r) s[r] = 0.f;
      const bf16* kbase = Kb + (size_t)(kb * 32 + t * 16) * HD_;
#pragma unroll
      for (int kk = 0; kk < 4; ++kk) {
        v16bf bf = load_b_frag(kbase + kk * 32, HD_, lane);
        s = wmma_bf16(qf[kk], bf, s);
      }
      const int keyhi = kb * 32 + t * 16 + 15;
      if (keyhi > q0) {                       // tile touches causal diagonal
        const int key = kb * 32 + t * 16 + (lane & 15);
#pragma unroll
        for (int r = 0; r < 8; ++r) {
          const int qr = q0 + r + 8 * half;
          s[r] = (key > qr) ? -1e30f : s[r] * scale;
        }
      } else {
#pragma unroll
        for (int r = 0; r < 8; ++r) s[r] *= scale;
      }
      sc[t] = s;
    }

    // online softmax: 16-lane butterfly reductions (wave32, xor masks < 16)
    float al[8];
#pragma unroll
    for (int r = 0; r < 8; ++r) {
      float rm = fmaxf(sc[0][r], sc[1][r]);
      rm = fmaxf(rm, __shfl_xor(rm, 8, 32));
      rm = fmaxf(rm, __shfl_xor(rm, 4, 32));
      rm = fmaxf(rm, __shfl_xor(rm, 2, 32));
      rm = fmaxf(rm, __shfl_xor(rm, 1, 32));
      const float mn = fmaxf(mrow[r], rm);
      al[r] = __expf(mrow[r] - mn);
      mrow[r] = mn;
      const float p0 = __expf(sc[0][r] - mn);
      const float p1 = __expf(sc[1][r] - mn);
      sc[0][r] = p0; sc[1][r] = p1;
      float rs = p0 + p1;
      rs += __shfl_xor(rs, 8, 32);
      rs += __shfl_xor(rs, 4, 32);
      rs += __shfl_xor(rs, 2, 32);
      rs += __shfl_xor(rs, 1, 32);
      lrow[r] = lrow[r] * al[r] + rs;
    }
#pragma unroll
    for (int c = 0; c < 8; ++c)
#pragma unroll
      for (int r = 0; r < 8; ++r) O[c][r] *= al[r];

    // Transpose P (accumulator layout -> A-fragment) via per-wave LDS.
#pragma unroll
    for (int t = 0; t < 2; ++t)
#pragma unroll
      for (int r = 0; r < 8; ++r) {
        const int row = r + 8 * half, colk = t * 16 + (lane & 15);
        pl[row * 32 + colk] = (bf16)sc[t][r];
      }
    asm volatile("s_wait_dscnt 0" ::: "memory");  // wave-local RAW fence
    v16bf pf = load_a_frag(pl, 32, lane);
    asm volatile("" ::: "memory");

    // O += P (16x32) x V (32x128); Vt rows are head-dim, contiguous in seq.
#pragma unroll
    for (int c = 0; c < 8; ++c) {
      v16bf vf = load_b_frag(Vb + (size_t)(c * 16) * S_ + kb * 32, S_, lane);
      O[c] = wmma_bf16(pf, vf, O[c]);
    }
  }

  // normalize and store context (bf16, [B][S][D] row-major)
#pragma unroll
  for (int c = 0; c < 8; ++c) {
    const int col = h * HD_ + c * 16 + (lane & 15);
#pragma unroll
    for (int r = 0; r < 8; ++r) {
      const int row = q0 + r + 8 * half;
      const float o = O[c][r] / lrow[r];
      ctx[((size_t)(b * S_ + row)) * D_ + col] = (bf16)o;
    }
  }
}

// ------------------------------------------------------ output projection
__global__ __launch_bounds__(256)
void out_proj(const bf16* __restrict__ Cx, const bf16* __restrict__ WO,
              float* __restrict__ out) {
  __shared__ bf16 As[128 * 32];
  __shared__ bf16 Bs[128 * 32];
  const int tileM = blockIdx.x * 128;
  const int tileN = blockIdx.y * 128;
  const int tid  = threadIdx.x;
  const int wave = tid >> 5, lane = tid & 31;
  const int wm = (wave & 3) * 32;
  const int wn = (wave >> 2) * 64;
  const int half = lane >> 4;

  v8f acc[2][4];
#pragma unroll
  for (int a = 0; a < 2; ++a)
#pragma unroll
    for (int b = 0; b < 4; ++b)
#pragma unroll
      for (int r = 0; r < 8; ++r) acc[a][b][r] = 0.f;

  for (int k0 = 0; k0 < D_; k0 += 32) {
#pragma unroll
    for (int part = 0; part < 2; ++part) {
      int idx = tid + part * 256;
      int row = idx >> 2, c = (idx & 3) * 8;
      *(v8bf*)(As + row * 32 + c) = *(const v8bf*)(Cx + (size_t)(tileM + row) * D_ + k0 + c);
      *(v8bf*)(Bs + row * 32 + c) = *(const v8bf*)(WO + (size_t)(tileN + row) * D_ + k0 + c);
    }
    __syncthreads();
    v16bf af0 = load_a_frag(As + (wm +  0) * 32, 32, lane);
    v16bf af1 = load_a_frag(As + (wm + 16) * 32, 32, lane);
#pragma unroll
    for (int nt = 0; nt < 4; ++nt) {
      v16bf bf = load_b_frag(Bs + (wn + nt * 16) * 32, 32, lane);
      acc[0][nt] = wmma_bf16(af0, bf, acc[0][nt]);
      acc[1][nt] = wmma_bf16(af1, bf, acc[1][nt]);
    }
    __syncthreads();
  }

#pragma unroll
  for (int mt = 0; mt < 2; ++mt)
#pragma unroll
    for (int nt = 0; nt < 4; ++nt) {
      const int col = tileN + wn + nt * 16 + (lane & 15);
#pragma unroll
      for (int r = 0; r < 8; ++r) {
        const int row = tileM + wm + mt * 16 + r + 8 * half;
        out[(size_t)row * D_ + col] = acc[mt][nt][r];
      }
    }
}

// ----------------------------------------------------------------- launch
extern "C" void kernel_launch(void* const* d_in, const int* in_sizes, int n_in,
                              void* d_out, int out_size, void* d_ws, size_t ws_size,
                              hipStream_t stream) {
  const float* x  = (const float*)d_in[0];
  const float* wq = (const float*)d_in[1];
  const float* wk = (const float*)d_in[2];
  const float* wv = (const float*)d_in[3];
  const float* wo = (const float*)d_in[4];

  char* ws = (char*)d_ws;
  size_t off = 0;
  bf16* xb  = (bf16*)(ws + off); off += (size_t)M_ * D_ * 2;  // 16 MiB
  bf16* wqb = (bf16*)(ws + off); off += (size_t)D_ * D_ * 2;  //  8 MiB
  bf16* wkb = (bf16*)(ws + off); off += (size_t)D_ * D_ * 2;
  bf16* wvb = (bf16*)(ws + off); off += (size_t)D_ * D_ * 2;
  bf16* wob = (bf16*)(ws + off); off += (size_t)D_ * D_ * 2;
  bf16* Qp  = (bf16*)(ws + off); off += (size_t)M_ * D_ * 2;  // [bh][s][hd]
  bf16* Kp  = (bf16*)(ws + off); off += (size_t)M_ * D_ * 2;  // [bh][s][hd]
  bf16* Vtp = (bf16*)(ws + off); off += (size_t)M_ * D_ * 2;  // [bh][hd][s]
  bf16* ctx = (bf16*)(ws + off); off += (size_t)M_ * D_ * 2;  // [b][s][d]
  (void)ws_size; (void)in_sizes; (void)n_in; (void)out_size;  // 112 MiB total

  auto cvt = [&](const float* src, bf16* dst, int n) {
    int n4 = n / 4;
    cvt_f32_bf16<<<(n4 + 255) / 256, 256, 0, stream>>>(src, dst, n4);
  };
  cvt(x,  xb,  M_ * D_);
  cvt(wq, wqb, D_ * D_);
  cvt(wk, wkb, D_ * D_);
  cvt(wv, wvb, D_ * D_);
  cvt(wo, wob, D_ * D_);

  qkv_rope_gemm<<<dim3(M_ / 128, D_ / 128, 3), 256, 0, stream>>>(
      xb, wqb, wkb, wvb, Qp, Kp, Vtp);
  flash_attn<<<dim3(S_ / 128, B_ * H_), 256, 0, stream>>>(Qp, Kp, Vtp, ctx);
  out_proj<<<dim3(M_ / 128, D_ / 128), 256, 0, stream>>>(ctx, wob, (float*)d_out);
}
